// PlainGCN_76974403879378
// MI455X (gfx1250) — matrix operand order repaired and checked by reading, not compiled, not run on hardware
//
#include <hip/hip_runtime.h>
#include <math.h>

// Problem constants (match reference)
#define NN   100000   // nodes
#define NE   300000   // edges
#define FEAT 3
#define HID  256
#define NCLS 2
#define NG   64

typedef long long i64;
typedef __attribute__((ext_vector_type(16))) __bf16 v16bf;
typedef __attribute__((ext_vector_type(8)))  __bf16 v8bf;
typedef __attribute__((ext_vector_type(8)))  float  v8f;
typedef unsigned int u32x4 __attribute__((ext_vector_type(4)));
typedef int          i32x4 __attribute__((ext_vector_type(4)));
typedef int          i32x8 __attribute__((ext_vector_type(8)));

// LDS row stride for staged W^T: 256 bf16 (512B) + 16B pad.
// 528B = 132 DWORDs; 132 % 64 = 4 -> 16 consecutive columns land on 16
// distinct bank groups -> conflict-free ds_load_b128 fragment reads.
#define LDSPAD 264

// ---------------------------------------------------------------- utility
__global__ void k_fill(float* p, float v, int n) {
    int i = blockIdx.x * 256 + threadIdx.x;
    if (i < n) p[i] = v;
}

// deg[dst] += 1 per edge (self-loop handled by init to 1.0)
__global__ void k_count_deg(const i64* __restrict__ dst, float* __restrict__ deg) {
    int e = blockIdx.x * 256 + threadIdx.x;
    if (e < NE) atomicAdd(&deg[(int)dst[e]], 1.0f);
}

// deg -> rsqrt(deg) in place (deg >= 1 always due to self loops)
__global__ void k_dinv(float* deg) {
    int i = blockIdx.x * 256 + threadIdx.x;
    if (i < NN) deg[i] = rsqrtf(deg[i]);
}

__global__ void k_edge_norm(const i64* __restrict__ src, const i64* __restrict__ dst,
                            const float* __restrict__ dinv, float* __restrict__ nrm) {
    int e = blockIdx.x * 256 + threadIdx.x;
    if (e < NE) nrm[e] = dinv[(int)src[e]] * dinv[(int)dst[e]];
}

// ---------------------------------------------------------------- layer 1 linear (K=3)
__global__ void k_lin1(const float* __restrict__ x, const float* __restrict__ W1,
                       float* __restrict__ Hlin) {
    int t = blockIdx.x * 256 + threadIdx.x;   // N*H threads
    if (t >= NN * HID) return;
    int n = t / HID, f = t % HID;
    float a = x[n * 3 + 0] * W1[0 * HID + f]
            + x[n * 3 + 1] * W1[1 * HID + f]
            + x[n * 3 + 2] * W1[2 * HID + f];
    Hlin[t] = a;
}

// ---------------------------------------------------------------- W (HxH f32 row-major) -> Wt (bf16, transposed)
__global__ void k_wt_bf16(const float* __restrict__ W, __bf16* __restrict__ Wt) {
    int t = blockIdx.x * 256 + threadIdx.x;   // H*H
    if (t >= HID * HID) return;
    int n = t / HID, k = t % HID;             // write coalesced along k
    Wt[t] = (__bf16)W[k * HID + n];
}

// ---------------------------------------------------------------- WMMA GEMM: Hlin = Hb(N x 256, bf16) @ W(256 x 256)
// Wt = W^T (bf16). One TDM tensor_load_to_lds stages all of Wt into LDS per
// block (with 16B/row padding for bank-conflict-free fragment reads); the
// K-loop then reads B fragments from LDS and streams A from global.
// Block = 256 threads = 8 waves; block tile = 32 rows x 256 cols.
// Wave (wm in 0..1, wn in 0..3): 16 rows x 64 cols = 4 accumulator tiles.
__global__ void __launch_bounds__(256)
k_gemm_bf16(const __bf16* __restrict__ Hb, const __bf16* __restrict__ Wt,
            float* __restrict__ out) {
    __shared__ __bf16 sB[256 * LDSPAD];   // 132 KB (<= 320 KB WGP LDS)

    const int lane = threadIdx.x & 31;
    const int wave = threadIdx.x >> 5;

#if __has_builtin(__builtin_amdgcn_tensor_load_to_lds)
    if (wave == 0) {
        // Tensor DMA descriptor (D#), CDNA5 ISA:
        unsigned long long ga = (unsigned long long)(const void*)Wt;
        unsigned int ldsOff = (unsigned int)(unsigned long long)(const void*)&sB[0];
        u32x4 g0 = {0, 0, 0, 0};
        g0[0] = 1u;                                             // count=1 (valid), user mode
        g0[1] = ldsOff;                                         // lds_addr
        g0[2] = (unsigned int)(ga & 0xFFFFFFFFull);             // global_addr[31:0]
        g0[3] = (unsigned int)((ga >> 32) & 0x1FFFFFFull)       // global_addr[56:32]
              | (2u << 30);                                     // type = 2 ("image")
        i32x8 g1 = {0, 0, 0, 0, 0, 0, 0, 0};
        g1[0] = (1 << 16)      // data_size = 2 bytes
              | (1 << 20)      // pad_enable (LDS destination padding)
              | (6 << 22)      // pad_interval: every 128 DWORDs (512 B row)
              | (3 << 25);     // pad_amount: 4 DWORDs (16 B)
        g1[1] = (int)(256u << 16);   // tensor_dim0[15:0] = 256 (bits 63:48)
        g1[2] = (int)(256u << 16);   // tensor_dim1[15:0] = 256 (bits 95:80)
        g1[3] = (int)(256u << 16);   // tile_dim0 = 256      (bits 127:112)
        g1[4] = 256;                 // tile_dim1 = 256      (bits 143:128)
        g1[5] = 256;                 // tensor_dim0_stride = 256 (bits 207:160)
        i32x4 gz = {0, 0, 0, 0};
#if defined(__clang_major__) && __clang_major__ >= 23
        i32x8 gz8 = {0, 0, 0, 0, 0, 0, 0, 0};
        __builtin_amdgcn_tensor_load_to_lds(g0, g1, gz, gz, gz8, 0);
#else
        __builtin_amdgcn_tensor_load_to_lds(g0, g1, gz, gz, 0);
#endif
        __builtin_amdgcn_s_wait_tensorcnt(0);
    }
    __syncthreads();
#else
    // Cooperative fallback: thread t copies column t (256 bf16 = 512 B)
    {
        int c = threadIdx.x;
        const __bf16* s = Wt + (long)c * HID;
        __bf16* d = sB + (long)c * LDSPAD;
        #pragma unroll
        for (int k = 0; k < HID; k += 8)
            *(v8bf*)(d + k) = *(const v8bf*)(s + k);
    }
    __syncthreads();
#endif

    const int wm   = wave >> 2;          // 0..1
    const int wn   = wave & 3;           // 0..3
    const int m0   = blockIdx.x * 32 + wm * 16;   // N % 32 == 0 -> no guard
    const int c0   = wn * 64;

    // A fragment addressing (16-bit A 16x32, this code's contiguous k order, same as B):
    //   lane<16: row=lane,    K [0..15]
    //   lane>=16: row=lane-16, K [16..31]
    const int rowA = m0 + (lane & 15);
    const int kA   = (lane < 16) ? 0 : 16;
    // B fragment addressing (16-bit B 32x16): lane = column, 16 consecutive K:
    //   lane<16: col=c+lane, K [0..15] ; lane>=16: col=c+lane-16, K [16..31]
    const int colB = c0 + (lane & 15);
    const int kB   = (lane < 16) ? 0 : 16;

    v8f acc0 = {}, acc1 = {}, acc2 = {}, acc3 = {};

    const __bf16* aBase = Hb + (long)rowA * HID + kA;
    const __bf16* bBase = sB + (long)colB * LDSPAD + kB;
    #pragma unroll
    for (int k0 = 0; k0 < HID; k0 += 32) {
        v16bf a = *(const v16bf*)(aBase + k0);

        v16bf b0 = *(const v16bf*)(bBase + k0);                 // from LDS
        v16bf b1 = *(const v16bf*)(bBase + 16 * LDSPAD + k0);
        v16bf b2 = *(const v16bf*)(bBase + 32 * LDSPAD + k0);
        v16bf b3 = *(const v16bf*)(bBase + 48 * LDSPAD + k0);

        acc0 = __builtin_amdgcn_wmma_f32_16x16x32_bf16(false, a, false, b0, (short)0, acc0, false, false);
        acc1 = __builtin_amdgcn_wmma_f32_16x16x32_bf16(false, a, false, b1, (short)0, acc1, false, false);
        acc2 = __builtin_amdgcn_wmma_f32_16x16x32_bf16(false, a, false, b2, (short)0, acc2, false, false);
        acc3 = __builtin_amdgcn_wmma_f32_16x16x32_bf16(false, a, false, b3, (short)0, acc3, false, false);
    }

    // D layout (f32 16x16): lane l, vgpr v -> D[M = v + 8*(l>=16)][Ncol = l&15]
    const int rbase = m0 + ((lane >> 4) << 3);
    const int col   = c0 + (lane & 15);
    float* o = out + (long)rbase * HID + col;
    #pragma unroll
    for (int v = 0; v < 8; ++v) {
        o[(long)v * HID +  0] = acc0[v];
        o[(long)v * HID + 16] = acc1[v];
        o[(long)v * HID + 32] = acc2[v];
        o[(long)v * HID + 48] = acc3[v];
    }
}

// ---------------------------------------------------------------- Agg = dinv[n]^2 * Hlin[n] (self-loop term; initializes Agg)
__global__ void k_selfloop(const float* __restrict__ dinv, const float* __restrict__ Hlin,
                           float* __restrict__ Agg) {
    int t = blockIdx.x * 256 + threadIdx.x;   // N*H/4 threads
    if (t >= NN * (HID / 4)) return;
    int n  = t / (HID / 4);
    int f4 = (t % (HID / 4)) * 4;
    float w = dinv[n]; w *= w;
    float4 h = *(const float4*)(Hlin + (long)n * HID + f4);
    float4 o = make_float4(h.x * w, h.y * w, h.z * w, h.w * w);
    *(float4*)(Agg + (long)n * HID + f4) = o;
}

// ---------------------------------------------------------------- edge scatter: Agg[dst] += norm * Hlin[src]
__global__ void k_scatter(const i64* __restrict__ src, const i64* __restrict__ dst,
                          const float* __restrict__ nrm, const float* __restrict__ Hlin,
                          float* __restrict__ Agg) {
    int t = blockIdx.x * 256 + threadIdx.x;   // E * 64 threads, 4 features each
    int e = t >> 6;
    if (e >= NE) return;
    int f4 = (t & 63) * 4;
    int s = (int)src[e], d = (int)dst[e];
    float w = nrm[e];
    float4 h = *(const float4*)(Hlin + (long)s * HID + f4);
    float* a = Agg + (long)d * HID + f4;
    atomicAdd(a + 0, h.x * w);
    atomicAdd(a + 1, h.y * w);
    atomicAdd(a + 2, h.z * w);
    atomicAdd(a + 3, h.w * w);
}

// ---------------------------------------------------------------- Hb = bf16(relu(Agg + b))
__global__ void k_bias_relu_bf16(const float* __restrict__ Agg, const float* __restrict__ b,
                                 __bf16* __restrict__ Hb) {
    int t = blockIdx.x * 256 + threadIdx.x;   // N*H/4
    if (t >= NN * (HID / 4)) return;
    long base = (long)t * 4;
    int f4 = (int)(base % HID);
    float4 v = *(const float4*)(Agg + base);
    float4 bb = *(const float4*)(b + f4);
    Hb[base + 0] = (__bf16)fmaxf(v.x + bb.x, 0.0f);
    Hb[base + 1] = (__bf16)fmaxf(v.y + bb.y, 0.0f);
    Hb[base + 2] = (__bf16)fmaxf(v.z + bb.z, 0.0f);
    Hb[base + 3] = (__bf16)fmaxf(v.w + bb.w, 0.0f);
}

// ---------------------------------------------------------------- pooling
__global__ void k_count_batch(const i64* __restrict__ batch, float* __restrict__ cnt) {
    int n = blockIdx.x * 256 + threadIdx.x;
    if (n < NN) atomicAdd(&cnt[(int)batch[n]], 1.0f);
}

__global__ void k_pool(const i64* __restrict__ batch, const __bf16* __restrict__ Hb,
                       float* __restrict__ pooled) {
    int t = blockIdx.x * 256 + threadIdx.x;   // N * 64 threads
    int n = t >> 6;
    if (n >= NN) return;
    int f4 = (t & 63) * 4;
    int g = (int)batch[n];
    const __bf16* h = Hb + (long)n * HID + f4;
    float* p = pooled + (long)g * HID + f4;
    atomicAdd(p + 0, (float)h[0]);
    atomicAdd(p + 1, (float)h[1]);
    atomicAdd(p + 2, (float)h[2]);
    atomicAdd(p + 3, (float)h[3]);
}

__global__ void k_head(const float* __restrict__ pooled, const float* __restrict__ cnt,
                       const float* __restrict__ Wout, const float* __restrict__ bout,
                       float* __restrict__ out) {
    int t = threadIdx.x;                      // NG*NCLS = 128 threads
    if (t >= NG * NCLS) return;
    int g = t / NCLS, c = t % NCLS;
    float inv = 1.0f / fmaxf(cnt[g], 1.0f);
    float acc = bout[c];
    for (int f = 0; f < HID; ++f)
        acc += pooled[(long)g * HID + f] * inv * Wout[f * NCLS + c];
    out[t] = 1.0f / (1.0f + expf(-acc));
}

// ---------------------------------------------------------------- launcher
extern "C" void kernel_launch(void* const* d_in, const int* in_sizes, int n_in,
                              void* d_out, int out_size, void* d_ws, size_t ws_size,
                              hipStream_t stream) {
    const float* x     = (const float*)d_in[0];
    const i64*   ei    = (const i64*)d_in[1];     // [2, E] int64
    const i64*   batch = (const i64*)d_in[2];     // [N] int64
    const float* W1    = (const float*)d_in[3];
    const float* b1    = (const float*)d_in[4];
    const float* Ws[4] = { (const float*)d_in[5], (const float*)d_in[7],
                           (const float*)d_in[9], (const float*)d_in[11] };
    const float* bs[4] = { (const float*)d_in[6], (const float*)d_in[8],
                           (const float*)d_in[10], (const float*)d_in[12] };
    const float* Wout  = (const float*)d_in[13];
    const float* bout  = (const float*)d_in[14];
    float* out = (float*)d_out;

    const i64* src = ei;
    const i64* dst = ei + NE;

    const long NH = (long)NN * HID;

    // workspace carve-up (all offsets keep >=32B alignment)
    float*  deg    = (float*)d_ws;                 // N
    float*  nrm    = deg + NN;                     // E
    float*  Hlin   = nrm + NE;                     // N*H
    float*  Agg    = Hlin + NH;                    // N*H
    float*  pooled = Agg + NH;                     // G*H
    float*  cnt    = pooled + (long)NG * HID;      // 64 (padded)
    __bf16* Hb     = (__bf16*)(cnt + 64);          // N*H bf16
    __bf16* Wt     = Hb + NH;                      // H*H bf16 (transposed weights)

    const int NB_N    = (NN + 255) / 256;
    const int NB_E    = (NE + 255) / 256;
    const int NB_NH   = (int)(NH / 256);           // 100000
    const int NB_NH4  = (int)(NH / 4 / 256);       // 25000
    const int NB_EDGE = NE * 64 / 256;             // 75000
    const int NB_POOL = NN * 64 / 256;             // 25000
    const int NB_GEMM = NN / 32;                   // 3125

    // degrees & edge norms
    k_fill<<<NB_N, 256, 0, stream>>>(deg, 1.0f, NN);              // self loop
    k_count_deg<<<NB_E, 256, 0, stream>>>(dst, deg);
    k_dinv<<<NB_N, 256, 0, stream>>>(deg);                        // deg -> dinv
    k_edge_norm<<<NB_E, 256, 0, stream>>>(src, dst, deg, nrm);

    // ---- layer 1: x @ W1 (K=3), aggregate, bias+relu -> Hb
    k_lin1<<<NB_NH, 256, 0, stream>>>(x, W1, Hlin);
    k_selfloop<<<NB_NH4, 256, 0, stream>>>(deg, Hlin, Agg);
    k_scatter<<<NB_EDGE, 256, 0, stream>>>(src, dst, nrm, Hlin, Agg);
    k_bias_relu_bf16<<<NB_NH4, 256, 0, stream>>>(Agg, b1, Hb);

    // ---- layers 2..5: WMMA GEMM (TDM-staged weights), aggregate, bias+relu
    for (int l = 0; l < 4; ++l) {
        k_wt_bf16<<<HID * HID / 256, 256, 0, stream>>>(Ws[l], Wt);
        k_gemm_bf16<<<NB_GEMM, 256, 0, stream>>>(Hb, Wt, Hlin);
        k_selfloop<<<NB_NH4, 256, 0, stream>>>(deg, Hlin, Agg);
        k_scatter<<<NB_EDGE, 256, 0, stream>>>(src, dst, nrm, Hlin, Agg);
        k_bias_relu_bf16<<<NB_NH4, 256, 0, stream>>>(Agg, bs[l], Hb);
    }

    // ---- global mean pool + head
    k_fill<<<(NG * HID + 64 + 255) / 256, 256, 0, stream>>>(pooled, 0.0f, NG * HID + 64);
    k_count_batch<<<NB_N, 256, 0, stream>>>(batch, cnt);
    k_pool<<<NB_POOL, 256, 0, stream>>>(batch, Hb, pooled);
    k_head<<<1, 128, 0, stream>>>(pooled, cnt, Wout, bout, out);
}
